// GVAET_18786186953237
// MI455X (gfx1250) — matrix operand order repaired and checked by reading, not compiled, not run on hardware
//
#include <hip/hip_runtime.h>
#include <math.h>

// Model dims
#define OBS_   8
#define PRED_  4
#define NAG_   256
#define FIN_   2
#define FOUT_  2
#define H_     128
#define NH_    8
#define DH_    16
#define FF_    2048
#define NL_    6
#define NNODE  2048          // OBS_*NAG_
#define SEQ_   256           // transformer sequence length (agents)
#define BD_    128           // transformer "batch" dim (old hidden channels)
#define ROWS_  32768         // SEQ_*BD_
#define SLAB_  2048          // FF row slab
#define NSLAB  16            // ROWS_/SLAB_

typedef __bf16 bf16;
typedef __attribute__((ext_vector_type(16))) __bf16 v16bf;
typedef __attribute__((ext_vector_type(8)))  __bf16 v8bf;
typedef __attribute__((ext_vector_type(8)))  float  v8f;

// A fragment (16x32 bf16 tile row): lane's data = 16 contiguous bf16 at +0
__device__ __forceinline__ v16bf load_a_bf(const bf16* __restrict__ p) {
  v8bf lo = *(const v8bf*)(p);
  v8bf hi = *(const v8bf*)(p + 8);
  return __builtin_shufflevector(lo, hi, 0, 1, 2, 3, 4, 5, 6, 7,
                                 8, 9, 10, 11, 12, 13, 14, 15);
}

// ---------------------------------------------------------------------------
// GEMM: C[m][n] = op( sum_k A[m][k]*B[n][k] + bias[n] )
// bf16 operands (k-contiguous), f32 accumulate. One wave computes a 32x64 C
// block: 2 A fragments x 4 B fragments -> 8 v_wmma_f32_16x16x32_bf16 / K-step.
// Inner loop is pure b128 loads + WMMA (no conversions).
// Optionally writes f32 C (Cf) and/or bf16 C (Cb) — fused output conversion.
// REQUIRES: M%32==0, N%64==0, K%32==0 (all call sites satisfy this).
// op: 0 = none, 1 = relu, 2 = min(.,10)
// ---------------------------------------------------------------------------
__global__ void k_gemm(const bf16* __restrict__ A, const bf16* __restrict__ B,
                       const float* __restrict__ bias, float* __restrict__ Cf,
                       bf16* __restrict__ Cb, int K, int lda, int ldb, int ldc, int op)
{
  const int lane = threadIdx.x;        // 0..31
  const int half = lane >> 4;          // 0 | 1
  const int l15  = lane & 15;
  const int n0   = blockIdx.x * 64;
  const int m0   = blockIdx.y * 32;

  const bf16* Ap0 = A + (m0 + l15) * lda + half * 16;   // lanes 16-31: K half  +16
  const bf16* Ap1 = Ap0 + 16 * lda;
  const bf16* Bp  = B + (n0 + l15) * ldb + half * 16;   // lanes 16-31: K half  +16

  v8f acc[2][4];
#pragma unroll
  for (int u = 0; u < 2; ++u)
#pragma unroll
    for (int t = 0; t < 4; ++t) acc[u][t] = v8f{};

  for (int k0 = 0; k0 < K; k0 += 32) {
    __builtin_prefetch(Ap0 + k0 + 64, 0, 0);            // global_prefetch_b8
    __builtin_prefetch(Bp + k0 + 64, 0, 0);
    v16bf a0 = load_a_bf(Ap0 + k0);
    v16bf a1 = load_a_bf(Ap1 + k0);
    v16bf b[4];
#pragma unroll
    for (int t = 0; t < 4; ++t) b[t] = *(const v16bf*)(Bp + t * 16 * ldb + k0);
#pragma unroll
    for (int t = 0; t < 4; ++t)
      acc[0][t] = __builtin_amdgcn_wmma_f32_16x16x32_bf16(false, a0, false, b[t],
                                                          (short)0, acc[0][t], false, false);
#pragma unroll
    for (int t = 0; t < 4; ++t)
      acc[1][t] = __builtin_amdgcn_wmma_f32_16x16x32_bf16(false, a1, false, b[t],
                                                          (short)0, acc[1][t], false, false);
  }

#pragma unroll
  for (int u = 0; u < 2; ++u) {
#pragma unroll
    for (int t = 0; t < 4; ++t) {
      int n = n0 + t * 16 + l15;
      float bz = (bias != nullptr) ? bias[n] : 0.0f;
#pragma unroll
      for (int r = 0; r < 8; ++r) {
        int m = m0 + u * 16 + r + half * 8;  // C/D layout: vgpr r, lane-half -> M+8
        float v = acc[u][t][r] + bz;
        if (op == 1)      v = fmaxf(v, 0.0f);
        else if (op == 2) v = fminf(v, 10.0f);
        if (Cf != nullptr) Cf[m * ldc + n] = v;
        if (Cb != nullptr) Cb[m * ldc + n] = (bf16)v;
      }
    }
  }
}

// ---------------------------------------------------------------------------
// logits = Z @ Z^T (2048x2048, K=128) fused with struct-loss epilogue.
// Same 32x64-per-wave WMMA scheme; per-block partials, fixed-order reduce.
// ---------------------------------------------------------------------------
__global__ void k_struct(const bf16* __restrict__ Z, const float* __restrict__ ADJ,
                         float* __restrict__ part)
{
  const int lane = threadIdx.x;
  const int half = lane >> 4;
  const int l15  = lane & 15;
  const int n0   = blockIdx.x * 64;
  const int m0   = blockIdx.y * 32;

  const bf16* Ap0 = Z + (m0 + l15) * H_ + half * 16;
  const bf16* Ap1 = Ap0 + 16 * H_;
  const bf16* Bp  = Z + (n0 + l15) * H_ + half * 16;

  v8f acc[2][4];
#pragma unroll
  for (int u = 0; u < 2; ++u)
#pragma unroll
    for (int t = 0; t < 4; ++t) acc[u][t] = v8f{};

#pragma unroll
  for (int k0 = 0; k0 < H_; k0 += 32) {
    v16bf a0 = load_a_bf(Ap0 + k0);
    v16bf a1 = load_a_bf(Ap1 + k0);
    v16bf b[4];
#pragma unroll
    for (int t = 0; t < 4; ++t) b[t] = *(const v16bf*)(Bp + t * 16 * H_ + k0);
#pragma unroll
    for (int t = 0; t < 4; ++t)
      acc[0][t] = __builtin_amdgcn_wmma_f32_16x16x32_bf16(false, a0, false, b[t],
                                                          (short)0, acc[0][t], false, false);
#pragma unroll
    for (int t = 0; t < 4; ++t)
      acc[1][t] = __builtin_amdgcn_wmma_f32_16x16x32_bf16(false, a1, false, b[t],
                                                          (short)0, acc[1][t], false, false);
  }

  float lsum = 0.0f;
#pragma unroll
  for (int u = 0; u < 2; ++u) {
#pragma unroll
    for (int t = 0; t < 4; ++t) {
      int n = n0 + t * 16 + l15;
#pragma unroll
      for (int r = 0; r < 8; ++r) {
        int m = m0 + u * 16 + r + half * 8;
        float lg  = acc[u][t][r];
        float adj = ADJ[m * NNODE + n];
        lsum += fmaxf(lg, 0.0f) - lg * adj + log1pf(expf(-fabsf(lg)));
      }
    }
  }
  __shared__ float red[32];
  red[lane] = lsum;
  __syncthreads();
  for (int off = 16; off; off >>= 1) {
    if (lane < off) red[lane] += red[lane + off];
    __syncthreads();
  }
  if (lane == 0) part[blockIdx.y * gridDim.x + blockIdx.x] = red[0];
}

// ------------------------- conversions / transpose -------------------------
__global__ void k_cvt(const float* __restrict__ src, bf16* __restrict__ dst, int n)
{
  int i = blockIdx.x * 256 + threadIdx.x;
  if (i < n) dst[i] = (bf16)src[i];
}

// (NNODE x H_) f32 -> (H_ x NNODE) bf16, so the agg GEMM's B is k-contiguous
__global__ void k_transpose(const float* __restrict__ src, bf16* __restrict__ dst)
{
  int idx = blockIdx.x * 256 + threadIdx.x;   // H_*NNODE
  int n = idx >> 11, k = idx & 2047;
  dst[idx] = (bf16)src[k * H_ + n];
}

// --------------------------- adjacency pipeline ----------------------------
__global__ void k_minmax_part(const float* __restrict__ a, int n,
                              float* __restrict__ pmn, float* __restrict__ pmx)
{
  int t = threadIdx.x;
  float mn = 3.0e38f, mx = -3.0e38f;
  for (int i = blockIdx.x * 256 + t; i < n; i += gridDim.x * 256) {
    float v = a[i];
    mn = fminf(mn, v); mx = fmaxf(mx, v);
  }
  __shared__ float smn[256], smx[256];
  smn[t] = mn; smx[t] = mx; __syncthreads();
  for (int off = 128; off; off >>= 1) {
    if (t < off) { smn[t] = fminf(smn[t], smn[t + off]); smx[t] = fmaxf(smx[t], smx[t + off]); }
    __syncthreads();
  }
  if (t == 0) { pmn[blockIdx.x] = smn[0]; pmx[blockIdx.x] = smx[0]; }
}

__global__ void k_minmax_final(const float* __restrict__ pmn, const float* __restrict__ pmx,
                               int n, float* __restrict__ stats)
{
  int t = threadIdx.x;
  float mn = 0.0f, mx = 0.0f;   // block-diag off-block entries are exactly 0
  for (int i = t; i < n; i += 256) { mn = fminf(mn, pmn[i]); mx = fmaxf(mx, pmx[i]); }
  __shared__ float smn[256], smx[256];
  smn[t] = mn; smx[t] = mx; __syncthreads();
  for (int off = 128; off; off >>= 1) {
    if (t < off) { smn[t] = fminf(smn[t], smn[t + off]); smx[t] = fmaxf(smx[t], smx[t + off]); }
    __syncthreads();
  }
  if (t == 0) { stats[0] = smn[0]; stats[1] = smx[0]; }
}

__global__ void k_build_adj(const float* __restrict__ adj_in, const float* __restrict__ stats,
                            float* __restrict__ ADJ)
{
  int idx = blockIdx.x * 256 + threadIdx.x;   // over NNODE*NNODE (fits int)
  int i = idx >> 11, j = idx & 2047;
  int ti = i >> 8, tj = j >> 8;
  float a = (ti == tj) ? adj_in[(ti << 16) + ((i & 255) << 8) + (j & 255)] : 0.0f;
  float mn = stats[0], mx = stats[1];
  ADJ[idx] = (a - mn) / (mx - mn);
}

__global__ void k_dis(const float* __restrict__ ADJ, float* __restrict__ dis)
{
  int j = blockIdx.x, t = threadIdx.x;
  float s = 0.0f;
  for (int i = t; i < NNODE; i += 256) {
    float v = ADJ[i * NNODE + j];
    if (i == j && ADJ[j * NNODE + j] == 0.0f) v += 1.0f;  // self-loop
    s += v;
  }
  __shared__ float red[256];
  red[t] = s; __syncthreads();
  for (int off = 128; off; off >>= 1) { if (t < off) red[t] += red[t + off]; __syncthreads(); }
  if (t == 0) { float dg = red[0]; dis[j] = (dg > 0.0f) ? (1.0f / sqrtf(dg)) : 0.0f; }
}

__global__ void k_build_agg(const float* __restrict__ ADJ, const float* __restrict__ dis,
                            bf16* __restrict__ AGG)
{
  int idx = blockIdx.x * 256 + threadIdx.x;
  int i = idx >> 11, j = idx & 2047;
  float v = ADJ[j * NNODE + i];                   // transpose of normalized A_sl
  if (i == j && ADJ[i * NNODE + i] == 0.0f) v += 1.0f;
  AGG[idx] = (bf16)(dis[j] * v * dis[i]);
}

// --------------------------- small projections -----------------------------
__global__ void k_enc1(const float* __restrict__ traj_in, const float* __restrict__ w,
                       const float* __restrict__ b, bf16* __restrict__ h1)
{
  int idx = blockIdx.x * 256 + threadIdx.x;             // NNODE*H_
  int row = idx >> 7, j = idx & 127;
  float x0 = traj_in[row * FIN_ + 0];
  float x1 = traj_in[row * FIN_ + 1];
  float v = b[j] + x0 * w[j * 2 + 0] + x1 * w[j * 2 + 1];
  h1[idx] = (bf16)fmaxf(v, 0.0f);
}

__global__ void k_z_kld(const float* __restrict__ mu, const float* __restrict__ ls,
                        const float* __restrict__ eps, const int* __restrict__ iftrain,
                        bf16* __restrict__ Z, float* __restrict__ part)
{
  int idx = blockIdx.x * 256 + threadIdx.x;             // NNODE*H_
  float m = mu[idx], l = ls[idx];
  Z[idx] = (bf16)((*iftrain > 0) ? (m + eps[idx] * expf(l)) : m);
  float term = 1.0f + 2.0f * l - m * m - expf(2.0f * l);
  __shared__ float red[256];
  int t = threadIdx.x;
  red[t] = term; __syncthreads();
  for (int off = 128; off; off >>= 1) { if (t < off) red[t] += red[t + off]; __syncthreads(); }
  if (t == 0) part[blockIdx.x] = red[0];
}

__global__ void k_x0(const float* __restrict__ gout, const float* __restrict__ w,
                     const float* __restrict__ b, float* __restrict__ x,
                     bf16* __restrict__ xbf)
{
  int idx = blockIdx.x * 256 + threadIdx.x;             // SEQ_*BD_*H_ = (s,b,h)
  int h = idx & 127, bb = (idx >> 7) & 127, s = idx >> 14;
  float acc = b[h];
#pragma unroll
  for (int t = 0; t < OBS_; ++t)
    acc += gout[(t * NAG_ + s) * H_ + bb] * w[h * OBS_ + t];
  x[idx] = acc;
  xbf[idx] = (bf16)acc;
}

// ------------------------------- attention ---------------------------------
// One block per (b, head); thread = query row s. K/V staged in LDS, online softmax.
__global__ void k_attn(const float* __restrict__ qkv, bf16* __restrict__ o)
{
  int bh = blockIdx.x;
  int b = bh >> 3, h = bh & 7;
  int s = threadIdx.x;                                  // 0..255
  __shared__ float ks[SEQ_][DH_];
  __shared__ float vs[SEQ_][DH_];
  int base = (s * BD_ + b) * 384 + h * DH_;
#pragma unroll
  for (int d = 0; d < DH_; ++d) {
    ks[s][d] = qkv[base + 128 + d];
    vs[s][d] = qkv[base + 256 + d];
  }
  float q[DH_];
#pragma unroll
  for (int d = 0; d < DH_; ++d) q[d] = qkv[base + d];
  __syncthreads();

  float mx = -3.0e38f, den = 0.0f, accv[DH_];
#pragma unroll
  for (int d = 0; d < DH_; ++d) accv[d] = 0.0f;
  for (int t = 0; t < SEQ_; ++t) {
    float sc = 0.0f;
#pragma unroll
    for (int d = 0; d < DH_; ++d) sc += q[d] * ks[t][d];
    sc *= 0.25f;                                        // 1/sqrt(16)
    float nm = fmaxf(mx, sc);
    float corr = expf(mx - nm);
    float wgt = expf(sc - nm);
    den = den * corr + wgt;
#pragma unroll
    for (int d = 0; d < DH_; ++d) accv[d] = accv[d] * corr + wgt * vs[t][d];
    mx = nm;
  }
  int obase = (s * BD_ + b) * H_ + h * DH_;
#pragma unroll
  for (int d = 0; d < DH_; ++d) o[obase + d] = (bf16)(accv[d] / den);
}

// ------------------------- residual + layernorm ----------------------------
__global__ void k_ln(float* __restrict__ x, const float* __restrict__ resid,
                     const float* __restrict__ g, const float* __restrict__ b,
                     bf16* __restrict__ xbf)
{
  int row = blockIdx.x;
  int t = threadIdx.x;                                  // 128
  __shared__ float red[128];
  __shared__ float mean_s, var_s;
  float v = x[row * H_ + t] + resid[row * H_ + t];
  red[t] = v; __syncthreads();
  for (int off = 64; off; off >>= 1) { if (t < off) red[t] += red[t + off]; __syncthreads(); }
  if (t == 0) mean_s = red[0] * (1.0f / H_);
  __syncthreads();
  float d = v - mean_s;
  red[t] = d * d; __syncthreads();
  for (int off = 64; off; off >>= 1) { if (t < off) red[t] += red[t + off]; __syncthreads(); }
  if (t == 0) var_s = red[0] * (1.0f / H_);
  __syncthreads();
  float outv = d / sqrtf(var_s + 1e-5f) * g[t] + b[t];
  x[row * H_ + t] = outv;
  xbf[row * H_ + t] = (bf16)outv;
}

// ------------------------------ output heads -------------------------------
__global__ void k_y1(const float* __restrict__ x, const float* __restrict__ w,
                     const float* __restrict__ b, float* __restrict__ y1)
{
  int idx = blockIdx.x * 256 + threadIdx.x;             // ROWS_*OBS_
  int row = idx >> 3, t8 = idx & 7;
  float acc = b[t8];
  for (int h = 0; h < H_; ++h) acc += x[row * H_ + h] * w[t8 * H_ + h];
  y1[idx] = acc;
}

__global__ void k_tra(const float* __restrict__ y1, const float* __restrict__ w,
                      const float* __restrict__ b, float* __restrict__ tra)
{
  int idx = blockIdx.x * 256 + threadIdx.x;             // OBS_*SEQ_*FOUT_ = 4096
  int t = idx >> 9, s = (idx >> 1) & 255, f = idx & 1;
  float acc = b[f];
  for (int bb = 0; bb < BD_; ++bb)
    acc += y1[(s * BD_ + bb) * OBS_ + t] * w[f * BD_ + bb];
  tra[idx] = acc;
}

__global__ void k_traj(const float* __restrict__ tra, const float* __restrict__ traj_in,
                       float* __restrict__ out, float* __restrict__ part)
{
  int idx = blockIdx.x * 256 + threadIdx.x;             // PRED_*SEQ_ = 1024
  int p = idx >> 8, s = idx & 255;
  float d2 = 0.0f;
#pragma unroll
  for (int f = 0; f < FOUT_; ++f) {
    float tv = tra[((PRED_ + p) * SEQ_ + s) * FOUT_ + f];
    float ty = traj_in[((OBS_ + p) * NAG_ + s) * FIN_ + f];
    out[p * SEQ_ * FOUT_ + s * FOUT_ + f] = tv;
    float d = tv - ty;
    d2 += d * d;
  }
  part[idx] = sqrtf(d2);
}

// ----------------------- deterministic scalar reduce -----------------------
__global__ void k_reduce(const float* __restrict__ src, int n, float scale,
                         float* __restrict__ dst)
{
  int t = threadIdx.x;                                  // single block, 256 threads
  float s = 0.0f;
  for (int i = t; i < n; i += 256) s += src[i];
  __shared__ float red[256];
  red[t] = s; __syncthreads();
  for (int off = 128; off; off >>= 1) { if (t < off) red[t] += red[t + off]; __syncthreads(); }
  if (t == 0) *dst = red[0] * scale;
}

__global__ void k_final(const float* __restrict__ scal, float* __restrict__ out)
{
  out[PRED_ * SEQ_ * FOUT_] = scal[0] + scal[1] + scal[2];   // kld + struct + traj
}

// ===========================================================================
extern "C" void kernel_launch(void* const* d_in, const int* in_sizes, int n_in,
                              void* d_out, int out_size, void* d_ws, size_t ws_size,
                              hipStream_t stream) {
  (void)in_sizes; (void)n_in; (void)out_size; (void)ws_size;

  const float* traj_in = (const float*)d_in[0];
  const float* adj_in  = (const float*)d_in[1];
  const float* eps     = (const float*)d_in[2];
  const float* pro1_w  = (const float*)d_in[3];
  const float* pro1_b  = (const float*)d_in[4];
  const float* pro2_w  = (const float*)d_in[5];
  const float* pro2_b  = (const float*)d_in[6];
  const float* mu_w    = (const float*)d_in[7];
  const float* mu_b    = (const float*)d_in[8];
  const float* ls_w    = (const float*)d_in[9];
  const float* ls_b    = (const float*)d_in[10];
  const float* gdl_w   = (const float*)d_in[11];
  const float* gdl_b   = (const float*)d_in[12];
  const float* pro3_w  = (const float*)d_in[13];
  const float* pro3_b  = (const float*)d_in[14];
  const float* pro4_w  = (const float*)d_in[15];
  const float* pro4_b  = (const float*)d_in[16];
  const float* pro5_w  = (const float*)d_in[17];
  const float* pro5_b  = (const float*)d_in[18];
  const float* qkv_w   = (const float*)d_in[19];
  const float* qkv_b   = (const float*)d_in[20];
  const float* out_w   = (const float*)d_in[21];
  const float* out_b   = (const float*)d_in[22];
  const float* ff1_w   = (const float*)d_in[23];
  const float* ff1_b   = (const float*)d_in[24];
  const float* ff2_w   = (const float*)d_in[25];
  const float* ff2_b   = (const float*)d_in[26];
  const float* ln1_g   = (const float*)d_in[27];
  const float* ln1_b   = (const float*)d_in[28];
  const float* ln2_g   = (const float*)d_in[29];
  const float* ln2_b   = (const float*)d_in[30];
  const int*   iftrain = (const int*)d_in[31];

  // ---- workspace carve-up (64B aligned); ~135 MiB total, L2-resident ----
  char* base = (char*)d_ws;
  size_t off = 0;
  auto alloc_f = [&](size_t n) { float* p = (float*)(base + off);
                                 off += ((n * 4 + 63) & ~(size_t)63); return p; };
  auto alloc_b = [&](size_t n) { bf16* p = (bf16*)(base + off);
                                 off += ((n * 2 + 63) & ~(size_t)63); return p; };

  float* ADJ    = alloc_f((size_t)NNODE * NNODE);
  float* dis    = alloc_f(NNODE);
  float* stats  = alloc_f(8);
  float* pmn    = alloc_f(512);
  float* pmx    = alloc_f(512);
  float* mulin  = alloc_f((size_t)NNODE * H_);
  float* lslin  = alloc_f((size_t)NNODE * H_);
  float* mu     = alloc_f((size_t)NNODE * H_);
  float* logstd = alloc_f((size_t)NNODE * H_);
  float* gout   = alloc_f((size_t)NNODE * H_);
  float* xb     = alloc_f((size_t)ROWS_ * H_);
  float* tmp    = alloc_f((size_t)ROWS_ * H_);
  float* qkvb   = alloc_f((size_t)ROWS_ * 384);
  float* y1     = alloc_f((size_t)ROWS_ * OBS_);
  float* tra    = alloc_f((size_t)OBS_ * SEQ_ * FOUT_);
  float* part   = alloc_f(16384);
  float* scal   = alloc_f(4);

  bf16* h1bf    = alloc_b((size_t)NNODE * H_);
  bf16* encbf   = alloc_b((size_t)NNODE * H_);
  bf16* mulinT  = alloc_b((size_t)NNODE * H_);
  bf16* lslinT  = alloc_b((size_t)NNODE * H_);
  bf16* Zbf     = alloc_b((size_t)NNODE * H_);
  bf16* AGGbf   = alloc_b((size_t)NNODE * NNODE);
  bf16* xbf     = alloc_b((size_t)ROWS_ * H_);
  bf16* hbuf    = alloc_b((size_t)SLAB_ * FF_);   // attn-out, then FF hidden
  bf16* w2bf    = alloc_b((size_t)H_ * H_);
  bf16* muwbf   = alloc_b((size_t)H_ * H_);
  bf16* lswbf   = alloc_b((size_t)H_ * H_);
  bf16* gdlwbf  = alloc_b((size_t)H_ * H_);
  bf16* qkvwbf  = alloc_b((size_t)NL_ * 3 * H_ * H_);
  bf16* outwbf  = alloc_b((size_t)NL_ * H_ * H_);
  bf16* ff1wbf  = alloc_b((size_t)NL_ * FF_ * H_);
  bf16* ff2wbf  = alloc_b((size_t)NL_ * H_ * FF_);

  auto cvt = [&](const float* s, bf16* d, int n) {
    k_cvt<<<(n + 255) / 256, 256, 0, stream>>>(s, d, n);
  };
  auto gemm = [&](const bf16* A, const bf16* B, const float* bias, float* Cf, bf16* Cb,
                  int M, int N, int K, int lda, int ldb, int ldc, int op) {
    dim3 g(N / 64, M / 32);   // N%64==0, M%32==0, K%32==0 at every call site
    k_gemm<<<g, dim3(32), 0, stream>>>(A, B, bias, Cf, Cb, K, lda, ldb, ldc, op);
  };

  // ---- one-shot weight conversion to bf16 (deterministic, graph-safe) ----
  cvt(pro2_w, w2bf,   H_ * H_);
  cvt(mu_w,   muwbf,  H_ * H_);
  cvt(ls_w,   lswbf,  H_ * H_);
  cvt(gdl_w,  gdlwbf, H_ * H_);
  cvt(qkv_w,  qkvwbf, NL_ * 3 * H_ * H_);
  cvt(out_w,  outwbf, NL_ * H_ * H_);
  cvt(ff1_w,  ff1wbf, NL_ * FF_ * H_);
  cvt(ff2_w,  ff2wbf, NL_ * H_ * FF_);

  // ---- adjacency: min/max, normalize, self-loops, sym-norm, transpose ----
  k_minmax_part<<<512, 256, 0, stream>>>(adj_in, OBS_ * NAG_ * NAG_, pmn, pmx);
  k_minmax_final<<<1, 256, 0, stream>>>(pmn, pmx, 512, stats);
  k_build_adj<<<(NNODE * NNODE) / 256, 256, 0, stream>>>(adj_in, stats, ADJ);
  k_dis<<<NNODE, 256, 0, stream>>>(ADJ, dis);
  k_build_agg<<<(NNODE * NNODE) / 256, 256, 0, stream>>>(ADJ, dis, AGGbf);

  // ---- encoder / VAE head ----
  k_enc1<<<(NNODE * H_) / 256, 256, 0, stream>>>(traj_in, pro1_w, pro1_b, h1bf);
  gemm(h1bf,  w2bf,  pro2_b, nullptr, encbf, NNODE, H_, H_, H_, H_, H_, 0);
  gemm(encbf, muwbf, mu_b,   mulin, nullptr, NNODE, H_, H_, H_, H_, H_, 0);
  gemm(encbf, lswbf, ls_b,   lslin, nullptr, NNODE, H_, H_, H_, H_, H_, 0);
  k_transpose<<<(NNODE * H_) / 256, 256, 0, stream>>>(mulin, mulinT);
  k_transpose<<<(NNODE * H_) / 256, 256, 0, stream>>>(lslin, lslinT);
  gemm(AGGbf, mulinT, nullptr, mu,     nullptr, NNODE, H_, NNODE, NNODE, NNODE, H_, 0);
  gemm(AGGbf, lslinT, nullptr, logstd, nullptr, NNODE, H_, NNODE, NNODE, NNODE, H_, 2);

  k_z_kld<<<(NNODE * H_) / 256, 256, 0, stream>>>(mu, logstd, eps, iftrain, Zbf, part);
  k_reduce<<<1, 256, 0, stream>>>(part, (NNODE * H_) / 256, -0.5f / (float)NNODE, scal + 0);

  gemm(Zbf, gdlwbf, gdl_b, gout, nullptr, NNODE, H_, H_, H_, H_, H_, 0);

  // logits = Z @ Z^T fused with struct loss (never materialized)
  k_struct<<<dim3(NNODE / 64, NNODE / 32), 32, 0, stream>>>(Zbf, ADJ, part);
  k_reduce<<<1, 256, 0, stream>>>(part, (NNODE / 64) * (NNODE / 32),
                                  1.0f / ((float)NNODE * (float)NNODE), scal + 1);

  // ---- into transformer space: x[s,b,h] ----
  k_x0<<<(SEQ_ * BD_ * H_) / 256, 256, 0, stream>>>(gout, pro3_w, pro3_b, xb, xbf);

  // ---- transformer layers ----
  for (int l = 0; l < NL_; ++l) {
    gemm(xbf, qkvwbf + (size_t)l * 3 * H_ * H_, qkv_b + (size_t)l * 3 * H_,
         qkvb, nullptr, ROWS_, 3 * H_, H_, H_, H_, 3 * H_, 0);
    k_attn<<<BD_ * NH_, SEQ_, 0, stream>>>(qkvb, hbuf);
    gemm(hbuf, outwbf + (size_t)l * H_ * H_, out_b + (size_t)l * H_,
         tmp, nullptr, ROWS_, H_, H_, H_, H_, H_, 0);
    k_ln<<<ROWS_, H_, 0, stream>>>(xb, tmp, ln1_g + (size_t)l * H_, ln1_b + (size_t)l * H_, xbf);

    for (int s = 0; s < NSLAB; ++s) {
      gemm(xbf + (size_t)s * SLAB_ * H_, ff1wbf + (size_t)l * FF_ * H_,
           ff1_b + (size_t)l * FF_, nullptr, hbuf,
           SLAB_, FF_, H_, H_, H_, FF_, 1);                             // relu
      gemm(hbuf, ff2wbf + (size_t)l * H_ * FF_, ff2_b + (size_t)l * H_,
           tmp + (size_t)s * SLAB_ * H_, nullptr,
           SLAB_, H_, FF_, FF_, FF_, H_, 0);
    }
    k_ln<<<ROWS_, H_, 0, stream>>>(xb, tmp, ln2_g + (size_t)l * H_, ln2_b + (size_t)l * H_, xbf);
  }

  // ---- output heads + losses ----
  k_y1<<<(ROWS_ * OBS_) / 256, 256, 0, stream>>>(xb, pro4_w, pro4_b, y1);
  k_tra<<<(OBS_ * SEQ_ * FOUT_) / 256, 256, 0, stream>>>(y1, pro5_w, pro5_b, tra);
  k_traj<<<(PRED_ * SEQ_) / 256, 256, 0, stream>>>(tra, traj_in, (float*)d_out, part);
  k_reduce<<<1, 256, 0, stream>>>(part, PRED_ * SEQ_, 1.0f / (float)SEQ_, scal + 2);
  k_final<<<1, 1, 0, stream>>>(scal, (float*)d_out);
}
